// HybridLatticeEngine_18846316494989
// MI455X (gfx1250) — matrix operand order repaired and checked by reading, not compiled, not run on hardware
//
#include <hip/hip_runtime.h>
#include <stdint.h>

// ---------------------------------------------------------------------------
// Problem constants (match reference: B=4, S=2048, D_IN=4096, D_OUT=4096, CAP=1024)
// ---------------------------------------------------------------------------
#define MDIM 8192   // B*S
#define KDIM 4096   // D_IN
#define NDIM 4096   // D_OUT
#define CAP  1024

#define BM 128      // workgroup tile M
#define BN 128      // workgroup tile N
#define BK 32       // k-step (one WMMA K)
#define NSTEPS (KDIM / BK)
#define LDS_STRIDE 40   // 32 + 8 bf16 pad (80 B row stride, 16B-aligned)

typedef __attribute__((ext_vector_type(16))) __bf16 v16bf;
typedef __attribute__((ext_vector_type(8)))  __bf16 v8bf;
typedef __attribute__((ext_vector_type(8)))  float  v8f;
typedef __attribute__((ext_vector_type(4)))  float  v4f;

// ---------------------------------------------------------------------------
// Truncation-based hi/lo bf16 split: hi = trunc(f), lo = trunc(f - hi).
// hi's truncation error is captured exactly by lo -> ~2^-16 combined rel err.
// ---------------------------------------------------------------------------
static __device__ __forceinline__ void split_bf16(float v, __bf16& h, __bf16& l) {
    union { float f; uint32_t u; } a; a.f = v;
    union { uint32_t u; float f; } hf; hf.u = a.u & 0xffff0000u;
    union { uint16_t u; __bf16 b; } hb; hb.u = (uint16_t)(a.u >> 16);
    h = hb.b;
    union { float f; uint32_t u; } r; r.f = v - hf.f;
    union { uint16_t u; __bf16 b; } lb; lb.u = (uint16_t)(r.u >> 16);
    l = lb.b;
}

// ---------------------------------------------------------------------------
// Kernel 1: logits[c] = dot(mem_keys[c,:], x[0,0,:])
// ---------------------------------------------------------------------------
__global__ __launch_bounds__(256)
void hle_logits(const float* __restrict__ keys, const float* __restrict__ x,
                float* __restrict__ logits) {
    __shared__ float red[256];
    const int t = threadIdx.x;
    const int c = blockIdx.x;
    const float* kr = keys + (size_t)c * KDIM;
    float p = 0.0f;
#pragma unroll
    for (int i = 0; i < KDIM / 256; ++i) {
        const int k = i * 256 + t;
        p += kr[k] * x[k];
    }
    red[t] = p;
    __syncthreads();
#pragma unroll
    for (int s = 128; s > 0; s >>= 1) {
        if (t < s) red[t] += red[t + s];
        __syncthreads();
    }
    if (t == 0) logits[c] = red[0];
}

// ---------------------------------------------------------------------------
// Kernel 2: attn = softmax(logits)   (single 1024-thread block)
// ---------------------------------------------------------------------------
__global__ __launch_bounds__(1024)
void hle_softmax(const float* __restrict__ logits, float* __restrict__ attn) {
    __shared__ float red[CAP];
    const int t = threadIdx.x;
    const float l = logits[t];
    red[t] = l;
    __syncthreads();
#pragma unroll
    for (int s = 512; s > 0; s >>= 1) {
        if (t < s) red[t] = fmaxf(red[t], red[t + s]);
        __syncthreads();
    }
    const float m = red[0];
    __syncthreads();
    const float e = __expf(l - m);
    red[t] = e;
    __syncthreads();
#pragma unroll
    for (int s = 512; s > 0; s >>= 1) {
        if (t < s) red[t] += red[t + s];
        __syncthreads();
    }
    attn[t] = e * (1.0f / red[0]);
}

// ---------------------------------------------------------------------------
// Kernel 3: memo[o] = 0.01 * sum_c attn[c] * mem_values[c,o]
// ---------------------------------------------------------------------------
__global__ __launch_bounds__(256)
void hle_memout(const float* __restrict__ attn, const float* __restrict__ vals,
                float* __restrict__ memo) {
    __shared__ float a[CAP];
    const int t = threadIdx.x;
#pragma unroll
    for (int i = t; i < CAP; i += 256) a[i] = attn[i];
    __syncthreads();
    const int o = blockIdx.x * 256 + t;
    float sum = 0.0f;
    for (int c = 0; c < CAP; ++c) sum += a[c] * vals[(size_t)c * NDIM + o];
    memo[o] = 0.01f * sum;
}

// ---------------------------------------------------------------------------
// Fragment gather from LDS.
// A frag (16x32): lane l: row = base + (l&15); K elems {h*8..h*8+7, 16+h*8..}
// B frag (32x16): lane l: row = base + (l&15); K elems {h*8..h*8+7, 16+h*8..}
// ---------------------------------------------------------------------------
static __device__ __forceinline__ v16bf ld_frag_a(const __bf16* rowp, int half) {
    union { v16bf v; v8bf h[2]; } u;
    u.h[0] = *(const v8bf*)(rowp + half * 8);
    u.h[1] = *(const v8bf*)(rowp + 16 + half * 8);
    return u.v;
}
static __device__ __forceinline__ v16bf ld_frag_b(const __bf16* rowp, int half) {
    union { v16bf v; v8bf h[2]; } u;
    u.h[0] = *(const v8bf*)(rowp + half * 8);
    u.h[1] = *(const v8bf*)(rowp + 16 + half * 8);
    return u.v;
}

// ---------------------------------------------------------------------------
// Convert one thread's staged registers (16 A floats + 16 B floats) into
// bf16 hi/lo and store to the given LDS buffer set.
// ---------------------------------------------------------------------------
static __device__ __forceinline__ void convert_store(
        __bf16 (*Ahi)[LDS_STRIDE], __bf16 (*Alo)[LDS_STRIDE],
        __bf16 (*Bhi)[LDS_STRIDE], __bf16 (*Blo)[LDS_STRIDE],
        int sRow, int sSeg, const v4f* aReg, const v4f* bReg) {
    union { v8bf v[2]; __bf16 e[16]; } ah, al, bh, bl;
#pragma unroll
    for (int g = 0; g < 4; ++g) {
#pragma unroll
        for (int e = 0; e < 4; ++e) {
            split_bf16(aReg[g][e], ah.e[g * 4 + e], al.e[g * 4 + e]);
            split_bf16(bReg[g][e], bh.e[g * 4 + e], bl.e[g * 4 + e]);
        }
    }
    *(v8bf*)(&Ahi[sRow][sSeg])     = ah.v[0];
    *(v8bf*)(&Ahi[sRow][sSeg + 8]) = ah.v[1];
    *(v8bf*)(&Alo[sRow][sSeg])     = al.v[0];
    *(v8bf*)(&Alo[sRow][sSeg + 8]) = al.v[1];
    *(v8bf*)(&Bhi[sRow][sSeg])     = bh.v[0];
    *(v8bf*)(&Bhi[sRow][sSeg + 8]) = bh.v[1];
    *(v8bf*)(&Blo[sRow][sSeg])     = bl.v[0];
    *(v8bf*)(&Blo[sRow][sSeg + 8]) = bl.v[1];
}

// ---------------------------------------------------------------------------
// Kernel 4: main GEMM, bf16x3 split through v_wmma_f32_16x16x32_bf16.
// Double-buffered LDS (80 KB of the WGP's 320 KB): 3-deep pipeline with ONE
// barrier per k-step:
//   iter i: WMMA from buf[i&1] | convert+store slab i+1 -> buf[(i+1)&1]
//           | global-load slab i+2 -> regs | barrier
// Block: 256 threads = 8 waves as 2(M) x 4(N); wave tile 64x32 = 4x2 WMMA tiles.
// Workgroup tile 128x128; grid = (N/128, M/128) = (32, 64).
// ---------------------------------------------------------------------------
__global__ __launch_bounds__(256)
void hle_gemm(const float* __restrict__ x, const float* __restrict__ lat,
              const float* __restrict__ scale, const float* __restrict__ memo,
              float* __restrict__ out) {
    __shared__ __bf16 sAhi[2][BM][LDS_STRIDE];
    __shared__ __bf16 sAlo[2][BM][LDS_STRIDE];
    __shared__ __bf16 sBhi[2][BN][LDS_STRIDE];
    __shared__ __bf16 sBlo[2][BN][LDS_STRIDE];

    const int tid  = threadIdx.x;
    const int lane = tid & 31;
    const int wid  = tid >> 5;
    const int half = lane >> 4;
    const int r    = lane & 15;

    // ---- staging assignment: 2 threads per tile row, 16 consecutive K each
    const int sRow = tid >> 1;          // 0..127
    const int sSeg = (tid & 1) * 16;    // 0 or 16
    const float* gA = x   + (size_t)(blockIdx.y * BM + sRow) * KDIM + sSeg;
    const float* gB = lat + (size_t)(blockIdx.x * BN + sRow) * KDIM + sSeg;

    // ---- wave tile position
    const int mBase = blockIdx.y * BM + (wid & 1) * 64;
    const int nBase = blockIdx.x * BN + (wid >> 1) * 32;
    const int mLoc  = (wid & 1) * 64;   // within LDS A tile
    const int nLoc  = (wid >> 1) * 32;  // within LDS B tile

    v8f zero = {};
    v8f acc[4][2];
#pragma unroll
    for (int mt = 0; mt < 4; ++mt)
#pragma unroll
        for (int nt = 0; nt < 2; ++nt) acc[mt][nt] = zero;

    // ---- prologue: slab 0 -> buf0; slab 1 staged in regs
    v4f aReg[4], bReg[4];
#pragma unroll
    for (int i = 0; i < 4; ++i) {
        aReg[i] = ((const v4f*)gA)[i];
        bReg[i] = ((const v4f*)gB)[i];
    }
    convert_store(sAhi[0], sAlo[0], sBhi[0], sBlo[0], sRow, sSeg, aReg, bReg);
#pragma unroll
    for (int i = 0; i < 4; ++i) {
        aReg[i] = ((const v4f*)(gA + BK))[i];
        bReg[i] = ((const v4f*)(gB + BK))[i];
    }
    __syncthreads();

    for (int ks = 0; ks < NSTEPS; ++ks) {
        const int cur = ks & 1;

        // ---- convert slab ks+1 (in regs) into the other buffer
        if (ks + 1 < NSTEPS) {
            convert_store(sAhi[cur ^ 1], sAlo[cur ^ 1],
                          sBhi[cur ^ 1], sBlo[cur ^ 1], sRow, sSeg, aReg, bReg);
        }
        // ---- issue global loads for slab ks+2 (overlaps the WMMA burst)
        if (ks + 2 < NSTEPS) {
            const float* pa = gA + (ks + 2) * BK;
            const float* pb = gB + (ks + 2) * BK;
#pragma unroll
            for (int i = 0; i < 4; ++i) {
                aReg[i] = ((const v4f*)pa)[i];
                bReg[i] = ((const v4f*)pb)[i];
            }
        }

        // ---- gather fragments from current buffer and run 24 WMMAs
        v16bf aHi[4], aLo[4], bHi[2], bLo[2];
#pragma unroll
        for (int mt = 0; mt < 4; ++mt) {
            aHi[mt] = ld_frag_a(&sAhi[cur][mLoc + mt * 16 + r][0], half);
            aLo[mt] = ld_frag_a(&sAlo[cur][mLoc + mt * 16 + r][0], half);
        }
#pragma unroll
        for (int nt = 0; nt < 2; ++nt) {
            bHi[nt] = ld_frag_b(&sBhi[cur][nLoc + nt * 16 + r][0], half);
            bLo[nt] = ld_frag_b(&sBlo[cur][nLoc + nt * 16 + r][0], half);
        }

#pragma unroll
        for (int mt = 0; mt < 4; ++mt) {
#pragma unroll
            for (int nt = 0; nt < 2; ++nt) {
                v8f c = acc[mt][nt];
                c = __builtin_amdgcn_wmma_f32_16x16x32_bf16(
                        false, aHi[mt], false, bHi[nt], (short)0, c, false, false);
                c = __builtin_amdgcn_wmma_f32_16x16x32_bf16(
                        false, aLo[mt], false, bHi[nt], (short)0, c, false, false);
                c = __builtin_amdgcn_wmma_f32_16x16x32_bf16(
                        false, aHi[mt], false, bLo[nt], (short)0, c, false, false);
                acc[mt][nt] = c;
            }
        }

        // one barrier per k-step: next iteration reads the buffer written above
        __syncthreads();
    }

    // ---- epilogue: fold per-channel dequant scale + memory retrieval term.
    // C/D layout: VGPR v -> M = v + half*8, lane&15 -> N column.
#pragma unroll
    for (int nt = 0; nt < 2; ++nt) {
        const int col = nBase + nt * 16 + r;
        const float s  = scale[col];
        const float mo = memo[col];
#pragma unroll
        for (int mt = 0; mt < 4; ++mt) {
#pragma unroll
            for (int v = 0; v < 8; ++v) {
                const int row = mBase + mt * 16 + half * 8 + v;
                out[(size_t)row * NDIM + col] = acc[mt][nt][v] * s + mo;
            }
        }
    }
}

// ---------------------------------------------------------------------------
// Host-side launcher
// ---------------------------------------------------------------------------
extern "C" void kernel_launch(void* const* d_in, const int* in_sizes, int n_in,
                              void* d_out, int out_size, void* d_ws, size_t ws_size,
                              hipStream_t stream) {
    (void)in_sizes; (void)n_in; (void)out_size; (void)ws_size;

    const float* x     = (const float*)d_in[0];   // [B,S,D_IN]
    const float* lat   = (const float*)d_in[1];   // [D_OUT, D_IN]
    const float* scale = (const float*)d_in[2];   // [D_OUT, 1]
    const float* keys  = (const float*)d_in[3];   // [CAP, D_IN]
    const float* vals  = (const float*)d_in[4];   // [CAP, D_OUT]
    float*       out   = (float*)d_out;           // [B,S,D_OUT]

    float* ws     = (float*)d_ws;
    float* logits = ws;               // [CAP]
    float* attn   = ws + CAP;         // [CAP]
    float* memo   = ws + 2 * CAP;     // [D_OUT] (pre-scaled by 0.01)

    hle_logits <<<CAP, 256, 0, stream>>>(keys, x, logits);
    hle_softmax<<<1, CAP, 0, stream>>>(logits, attn);
    hle_memout <<<NDIM / 256, 256, 0, stream>>>(attn, vals, memo);

    dim3 grid(NDIM / BN, MDIM / BM);   // (32, 64)
    hle_gemm<<<grid, 256, 0, stream>>>(x, lat, scale, memo, out);
}
